// Encoder_90409061581283
// MI455X (gfx1250) — matrix operand order repaired and checked by reading, not compiled, not run on hardware
//
#include <hip/hip_runtime.h>

#define B_  2
#define L_  1024
#define D_  768
#define H_  12
#define N_  256
#define W_  8
#define K_  50
#define NT_ 66
#define NR_ 57
#define NN_ 20
#define KK  (K_*K_)            /* 2500 pair rows */
#define MT  157                /* ceil(2500/16)  */

#define OFF_PRED (B_*KK*(1+NR_))     /* 290000 */
#define OFF_MASK (OFF_PRED + B_*KK)  /* 295000 */
#define OFF_TOP  (OFF_MASK + B_*KK)  /* 300000 */

typedef _Float16 v16h __attribute__((ext_vector_type(16)));
typedef _Float16 h8v  __attribute__((ext_vector_type(8)));
typedef float    v8f  __attribute__((ext_vector_type(8)));

// ---------------- 1: span embeddings + anchor scores (max/argmax) -----------
__global__ void k_span_embed(const float* __restrict__ seq,
                             const float* __restrict__ anchor,
                             const int* __restrict__ starts,
                             const int* __restrict__ lengths,
                             float* __restrict__ embs,
                             float* __restrict__ span_max,
                             int* __restrict__ cls) {
  int bn = blockIdx.x;              // b*N + n
  int b  = bn / N_;
  int t  = threadIdx.x;             // 256
  __shared__ float femb[D_];
  __shared__ float sc[NT_];
  int start = starts[bn];
  int len   = lengths[bn];          // in [0, W-1] -> valid count = len+1
  float wt  = 1.0f / (float)(len + 1);
  for (int d = t; d < D_; d += 256) {
    float acc = 0.f;
    for (int w = 0; w < W_; ++w) {
      if (w <= len) {
        int p = start + w; p = p < 0 ? 0 : (p > L_-1 ? L_-1 : p);
        acc += seq[((size_t)b * L_ + p) * D_ + d] * wt;
      }
    }
    femb[d] = acc;
    embs[(size_t)bn * D_ + d] = acc;
  }
  __syncthreads();
  if (t < NT_) {
    float acc = 0.f;
    const float* a = anchor + (size_t)t * D_;
    for (int d = 0; d < D_; ++d) acc += femb[d] * a[d];
    sc[t] = acc;
  }
  __syncthreads();
  if (t == 0) {
    float m = sc[0]; int mi = 0;
    for (int i = 1; i < NT_; ++i) if (sc[i] > m) { m = sc[i]; mi = i; }
    span_max[bn] = m;
    cls[bn] = mi;
  }
}

// ---------------- 2: top-K (sorted desc, tie -> lowest idx) + gathers -------
__global__ void k_topk(const float* __restrict__ span_max,
                       const int* __restrict__ cls,
                       const float* __restrict__ embs,
                       const unsigned char* __restrict__ is_trig,
                       int* __restrict__ top,
                       int* __restrict__ trig,
                       float* __restrict__ e,
                       float* __restrict__ out_top) {
  int b = blockIdx.x;
  int t = threadIdx.x;              // 256
  __shared__ float v[N_];
  __shared__ int idx[K_];
  for (int i = t; i < N_; i += 256) v[i] = span_max[b * N_ + i];
  __syncthreads();
  if (t == 0) {                     // tiny serial selection (N=256, K=50)
    for (int k = 0; k < K_; ++k) {
      float m = v[0]; int mi = 0;
      for (int i = 1; i < N_; ++i) if (v[i] > m) { m = v[i]; mi = i; }
      idx[k] = mi;
      v[mi] = -3.402823466e38f;
    }
  }
  __syncthreads();
  for (int k = 0; k < K_; ++k) {
    int src = idx[k];
    for (int d = t; d < D_; d += 256)
      e[((size_t)b * K_ + k) * D_ + d] = embs[((size_t)b * N_ + src) * D_ + d];
  }
  if (t < K_) {
    int src = idx[t];
    top[b * K_ + t]     = src;
    trig[b * K_ + t]    = is_trig[cls[b * N_ + src]] ? 1 : 0;
    out_top[b * K_ + t] = (float)src;
  }
}

// ---------------- 3: attention pooling only for selected top-K spans -------
__global__ void k_attpool(const float* __restrict__ att,
                          const int* __restrict__ starts,
                          const int* __restrict__ lengths,
                          const int* __restrict__ top,
                          float* __restrict__ a_e) {
  int bkh = blockIdx.x;             // (b*K + k)*H + h
  int h  = bkh % H_;
  int bk = bkh / H_;
  int b  = bk / K_;
  int n  = top[bk];
  int start = starts[b * N_ + n];
  int len   = lengths[b * N_ + n];
  float wt  = 1.0f / (float)(len + 1);
  int t = threadIdx.x;
  const float* abase = att + (size_t)(b * H_ + h) * L_ * L_;
  for (int l = t; l < L_; l += 256) {
    int lnext = l + 256;
    if (lnext < L_) __builtin_prefetch(abase + (size_t)start * L_ + lnext, 0, 1);
    float acc = 0.f;
    for (int w = 0; w <= len; ++w) {
      int p = start + w; p = p < 0 ? 0 : (p > L_-1 ? L_-1 : p);
      acc += abase[(size_t)p * L_ + l] * wt;
    }
    a_e[(size_t)bkh * L_ + l] = acc;
  }
}

// ---------------- 4: q = Gram over H, normalize, emit aw as f16 ------------
__global__ void k_q_aw(const float* __restrict__ a_e,
                       _Float16* __restrict__ aw_h) {
  int p  = blockIdx.x;              // b*KK + s*K + o
  int b  = p / KK;
  int so = p % KK;
  int s  = so / K_, o = so % K_;
  int t  = threadIdx.x;             // 256
  __shared__ float qv[L_];
  __shared__ float red[256];
  const float* As = a_e + (size_t)(b * K_ + s) * H_ * L_;
  const float* Ao = a_e + (size_t)(b * K_ + o) * H_ * L_;
  float psum = 0.f;
  for (int j = 0; j < L_/256; ++j) {
    int l = t + 256 * j;
    float acc = 0.f;
    for (int h = 0; h < H_; ++h) acc += As[h*L_ + l] * Ao[h*L_ + l];
    qv[l] = acc;
    psum += acc;
  }
  red[t] = psum;
  __syncthreads();
  for (int st = 128; st > 0; st >>= 1) {
    if (t < st) red[t] += red[t + st];
    __syncthreads();
  }
  float inv = 1.0f / red[0];
  for (int j = 0; j < L_/256; ++j) {
    int l = t + 256 * j;
    aw_h[((size_t)b * KK + so) * L_ + l] = (_Float16)(qv[l] * inv);
  }
}

// ---------------- 5: seq -> f16 --------------------------------------------
__global__ void k_seq2h(const float* __restrict__ seq, _Float16* __restrict__ seqh) {
  size_t i = (size_t)blockIdx.x * blockDim.x + threadIdx.x;
  if (i < (size_t)B_ * L_ * D_) seqh[i] = (_Float16)seq[i];
}

// ---------------- 6: c = aw @ seq  via v_wmma_f32_16x16x32_f16 -------------
// Block = 4 waves; one 16(M)x64(N) macro-tile per block, 16x16 per wave.
// A tile (16x32) staged once in LDS (shared by all 4 waves); B tile (32x64)
// staged TRANSPOSED in LDS so each lane's fragment is two aligned b128 reads.
// Fragment layouts per CDNA5 ISA:
//   A f16 16x32: lane<16 row M=lane holds K 0-7 (v0-3) & 16-23 (v4-7);
//                lane>=16 holds K 8-15 & 24-31.
//   B f16 32x16: lane<16 col N=lane holds K 0-7 & 16-23; lane>=16 holds K 8-15 & 24-31.
#define SA_ 40   /* LDS row stride (halves) for A: 32 + pad, 16B aligned */
#define SB_ 40   /* LDS col stride (halves) for B^T */
__global__ void __launch_bounds__(128) k_c_wmma(const _Float16* __restrict__ aw_h,
                                                const _Float16* __restrict__ seqh,
                                                float* __restrict__ c) {
  __shared__ _Float16 shA[16 * SA_];
  __shared__ _Float16 shB[64 * SB_];
  int t     = threadIdx.x;
  int lane  = t & 31;
  int wave  = t >> 5;
  int hi    = lane >> 4;                 // lane half select
  int mtile = blockIdx.x;                // 0..156
  int nbase = blockIdx.y * 64;           // 12 groups of 64 cols
  int b     = blockIdx.z;
  int colL  = wave * 16 + (lane & 15);   // block-local column 0..63
  int ncol  = nbase + colL;
  // A staging assignment (threads 0..63): row ar, 8-half chunk ak
  int ar    = t >> 2;
  int ak    = (t & 3) * 8;
  int arow  = mtile * 16 + ar;
  const _Float16* Asrc = aw_h + ((size_t)b * KK + (arow < KK ? arow : 0)) * L_;
  // B staging assignment (all 128 threads): k-row bk, 16-col chunk bn0
  int bk    = t >> 2;                    // 0..31
  int bn0   = (t & 3) * 16;              // 0,16,32,48
  const _Float16* Bbase = seqh + (size_t)b * L_ * D_ + nbase;
  v8f acc = {};
  #pragma unroll 2
  for (int kb = 0; kb < L_; kb += 32) {
    if (t < 64) {                        // stage A tile 16x32
      h8v av = {};
      if (arow < KK) av = *(const h8v*)(Asrc + kb + ak);
      *(h8v*)(shA + ar * SA_ + ak) = av;
    }
    {                                    // stage B tile 32x64, transposed
      const _Float16* src = Bbase + (size_t)(kb + bk) * D_ + bn0;
      h8v b0 = *(const h8v*)(src);
      h8v b1 = *(const h8v*)(src + 8);
      for (int i = 0; i < 8; ++i) shB[(bn0 + i)     * SB_ + bk] = b0[i];
      for (int i = 0; i < 8; ++i) shB[(bn0 + 8 + i) * SB_ + bk] = b1[i];
    }
    __syncthreads();
    v16h af, bf;
    {
      h8v a0  = *(const h8v*)(shA + (lane & 15) * SA_ + hi * 8);
      h8v a1  = *(const h8v*)(shA + (lane & 15) * SA_ + hi * 8 + 16);
      h8v bb0 = *(const h8v*)(shB + colL * SB_ + hi * 8);
      h8v bb1 = *(const h8v*)(shB + colL * SB_ + hi * 8 + 16);
      for (int i = 0; i < 8; ++i) {
        af[i] = a0[i];  af[i + 8] = a1[i];
        bf[i] = bb0[i]; bf[i + 8] = bb1[i];
      }
    }
    acc = __builtin_amdgcn_wmma_f32_16x16x32_f16(false, af, false, bf,
                                                 (short)0, acc, false, false);
    __syncthreads();
  }
  int mbase = mtile * 16 + hi * 8;       // C/D layout: vgpr i -> M=mbase+i
  for (int i = 0; i < 8; ++i) {
    int m = mbase + i;
    if (m < KK) c[((size_t)b * KK + m) * D_ + ncol] = acc[i];
  }
}

// ---------------- 7: e projections onto W1/W2/M1/M2 ------------------------
__global__ void k_eproj(const float* __restrict__ e,
                        const float* __restrict__ rel,
                        const float* __restrict__ nota,
                        float* __restrict__ eW1, float* __restrict__ eW2,
                        float* __restrict__ eM1, float* __restrict__ eM2) {
  int bk = blockIdx.x;              // B*K
  int t  = threadIdx.x;             // 128
  __shared__ float fe[D_];
  for (int d = t; d < D_; d += 128) fe[d] = e[(size_t)bk * D_ + d];
  __syncthreads();
  if (t < NR_) {
    const float* w1 = rel + (size_t)t * (3 * D_);
    const float* w2 = w1 + D_;
    float a1 = 0.f, a2 = 0.f;
    for (int d = 0; d < D_; ++d) { a1 += fe[d] * w1[d]; a2 += fe[d] * w2[d]; }
    eW1[bk * NR_ + t] = a1;
    eW2[bk * NR_ + t] = a2;
  } else if (t >= 64 && t < 64 + NN_) {
    int m = t - 64;
    const float* m1 = nota + (size_t)m * (3 * D_);
    const float* m2 = m1 + D_;
    float a1 = 0.f, a2 = 0.f;
    for (int d = 0; d < D_; ++d) { a1 += fe[d] * m1[d]; a2 += fe[d] * m2[d]; }
    eM1[bk * NN_ + m] = a1;
    eM2[bk * NN_ + m] = a2;
  }
}

// ---------------- 8: assemble scores / preds / pair_mask -------------------
__global__ void k_scores(const float* __restrict__ c,
                         const float* __restrict__ rel,
                         const float* __restrict__ nota,
                         const float* __restrict__ eW1, const float* __restrict__ eW2,
                         const float* __restrict__ eM1, const float* __restrict__ eM2,
                         const int* __restrict__ trig,
                         float* __restrict__ out) {
  int p  = blockIdx.x;              // b*KK + s*K + o
  int b  = p / KK;
  int so = p % KK;
  int s  = so / K_, o = so % K_;
  int t  = threadIdx.x;             // 128
  __shared__ float fc[D_];
  __shared__ float rsc[NR_];
  __shared__ float nsc[NN_];
  __shared__ float scv[1 + NR_];
  for (int d = t; d < D_; d += 128) fc[d] = c[(size_t)p * D_ + d];
  __syncthreads();
  if (t < NR_) {
    const float* w3 = rel + (size_t)t * (3 * D_) + 2 * D_;
    float a = 0.f; for (int d = 0; d < D_; ++d) a += fc[d] * w3[d];
    rsc[t] = a;
  } else if (t >= 64 && t < 64 + NN_) {
    int m = t - 64;
    const float* m3 = nota + (size_t)m * (3 * D_) + 2 * D_;
    float a = 0.f; for (int d = 0; d < D_; ++d) a += fc[d] * m3[d];
    nsc[m] = a;
  }
  __syncthreads();
  if (t == 0) {
    float nm = -3.402823466e38f;
    for (int m = 0; m < NN_; ++m) {
      float v = eM1[(b*K_+s)*NN_ + m] + eM2[(b*K_+o)*NN_ + m] + nsc[m];
      nm = v > nm ? v : nm;
    }
    scv[0] = nm;
  }
  if (t < NR_) scv[1 + t] = eW1[(b*K_+s)*NR_ + t] + eW2[(b*K_+o)*NR_ + t] + rsc[t];
  __syncthreads();
  bool masked = !((trig[b*K_ + s] != 0) && (s != o));
  for (int j = t; j < 1 + NR_; j += 128) {
    float v = masked ? 0.f : scv[j];
    out[(size_t)p * (1 + NR_) + j] = v;
    if (masked) scv[j] = 0.f;
  }
  __syncthreads();
  if (t == 0) {
    float m = scv[0]; int mi = 0;
    for (int j = 1; j < 1 + NR_; ++j) if (scv[j] > m) { m = scv[j]; mi = j; }
    out[OFF_PRED + p] = (float)mi;
    out[OFF_MASK + p] = masked ? 0.f : 1.f;
  }
}

// ---------------- host-side launcher ---------------------------------------
extern "C" void kernel_launch(void* const* d_in, const int* in_sizes, int n_in,
                              void* d_out, int out_size, void* d_ws, size_t ws_size,
                              hipStream_t stream) {
  const float* seq    = (const float*)d_in[0];
  const float* att    = (const float*)d_in[1];
  const float* anchor = (const float*)d_in[2];
  const float* rel    = (const float*)d_in[3];
  const float* nota   = (const float*)d_in[4];
  const int*   starts = (const int*)d_in[5];
  const int*   lens   = (const int*)d_in[6];
  const unsigned char* istrig = (const unsigned char*)d_in[7]; // jnp bool -> 1B
  float* out = (float*)d_out;

  char* ws = (char*)d_ws;
  size_t off = 0;
  auto carve = [&](size_t bytes) -> void* {
    off = (off + 255) & ~(size_t)255;
    void* p = ws + off;
    off += bytes;
    return p;
  };
  float*    embs  = (float*)   carve((size_t)B_*N_*D_*4);
  float*    smax  = (float*)   carve((size_t)B_*N_*4);
  int*      cls   = (int*)     carve((size_t)B_*N_*4);
  int*      top   = (int*)     carve((size_t)B_*K_*4);
  int*      trig  = (int*)     carve((size_t)B_*K_*4);
  float*    e     = (float*)   carve((size_t)B_*K_*D_*4);
  float*    eW1   = (float*)   carve((size_t)B_*K_*NR_*4);
  float*    eW2   = (float*)   carve((size_t)B_*K_*NR_*4);
  float*    eM1   = (float*)   carve((size_t)B_*K_*NN_*4);
  float*    eM2   = (float*)   carve((size_t)B_*K_*NN_*4);
  float*    a_e   = (float*)   carve((size_t)B_*K_*H_*L_*4);
  _Float16* aw_h  = (_Float16*)carve((size_t)B_*KK*L_*2);
  _Float16* seqh  = (_Float16*)carve((size_t)B_*L_*D_*2);
  float*    cbuf  = (float*)   carve((size_t)B_*KK*D_*4);

  k_span_embed<<<B_*N_, 256, 0, stream>>>(seq, anchor, starts, lens, embs, smax, cls);
  k_topk<<<B_, 256, 0, stream>>>(smax, cls, embs, istrig, top, trig, e, out + OFF_TOP);
  k_attpool<<<B_*K_*H_, 256, 0, stream>>>(att, starts, lens, top, a_e);
  k_q_aw<<<B_*KK, 256, 0, stream>>>(a_e, aw_h);
  k_seq2h<<<(B_*L_*D_ + 255) / 256, 256, 0, stream>>>(seq, seqh);
  k_c_wmma<<<dim3(MT, 12, B_), 128, 0, stream>>>(aw_h, seqh, cbuf);
  k_eproj<<<B_*K_, 128, 0, stream>>>(e, rel, nota, eW1, eW2, eM1, eM2);
  k_scores<<<B_*KK, 128, 0, stream>>>(cbuf, rel, nota, eW1, eW2, eM1, eM2, trig, out);
}
